// MaskedTransformerDecoderLayer_18004502905391
// MI455X (gfx1250) — matrix operand order repaired and checked by reading, not compiled, not run on hardware
//
#include <hip/hip_runtime.h>
#include <math.h>

// ---------------------------------------------------------------------------
// MaskedTransformerDecoderLayer for MI455X (gfx1250, wave32, WMMA).
// All GEMMs run on v_wmma_f32_16x16x32_bf16. fp32 tensors are converted to
// bf16 once; templated epilogues fuse bias / residual / GELU / sigmoid;
// softmax fuses the 1/sqrt(d) scale and log(prev_mask+1e-6) bias.
// bf16 is carried as raw unsigned short; loads/stores are forced into the
// global address space so they lower to global_load/store (LOADcnt only)
// instead of flat ops (LOADcnt+DScnt).
// ---------------------------------------------------------------------------

#define GLOBAL_AS __attribute__((address_space(1)))

typedef __attribute__((ext_vector_type(16))) __bf16 v16bf;
typedef __attribute__((ext_vector_type(8)))  float  v8f;
typedef __attribute__((ext_vector_type(4)))  unsigned int u32x4;

union frag_u { u32x4 u[2]; v16bf v; };

// round-to-nearest-even f32 -> bf16 (bits)
__device__ inline unsigned short f2bf(float f) {
  unsigned int u = __builtin_bit_cast(unsigned int, f);
  u += 0x7FFFu + ((u >> 16) & 1u);
  return (unsigned short)(u >> 16);
}
__device__ inline float bf2f(unsigned short b) {
  unsigned int u = ((unsigned int)b) << 16;
  return __builtin_bit_cast(float, u);
}

// A-matrix 16x32 bf16 fragment: lanes 0-15 hold K 0..15 contiguously in v0..7,
// lanes 16-31 hold K 16..31 (matches B fragment order).
__device__ inline v16bf load_a_frag(const unsigned short* rowptr, int sel) {
  frag_u t;
  t.u[0] = *(const GLOBAL_AS u32x4*)(rowptr + sel * 16);
  t.u[1] = *(const GLOBAL_AS u32x4*)(rowptr + 8 + sel * 16);
  return t.v;
}
// B-matrix 32x16 bf16 fragment: lane = column; lanes 0-15 hold K 0..15
// contiguously in v0..7, lanes 16-31 hold K 16..31.
__device__ inline v16bf load_b_frag(const unsigned short* colptr, int sel) {
  const GLOBAL_AS u32x4* p = (const GLOBAL_AS u32x4*)(colptr + sel * 16);
  frag_u t;
  t.u[0] = p[0];
  t.u[1] = p[1];
  return t.v;
}

// Generic batched GEMM:  C[m,n] = sum_k A[m,k] * B[n,k]   (B rows = out feats)
// batch z: zo = z/inner, zi = z%inner; per-operand offset zo*s?o + zi*s?i.
// Compile-time epilogue: (+bias[n]) (+resid[idxC]) ACT(0 none,1 GELU,2 sigmoid)
// STF: store fp32 to Cf; STB: store bf16 to Cb.
// Requires: M,N multiples of 64 (waves past the edge idle), K multiple of 32.
template <bool BIAS, bool RESID, int ACT, bool STF, bool STB>
__global__ __launch_bounds__(256)
void gemm_bf16_wmma(const unsigned short* __restrict__ A, long long lda,
                    long long sAo, long long sAi,
                    const unsigned short* __restrict__ B, long long ldb,
                    long long sBo, long long sBi,
                    float* __restrict__ Cf, unsigned short* __restrict__ Cb,
                    long long ldc, long long sCo, long long sCi,
                    int inner, int M, int N, int K,
                    const float* __restrict__ bias,
                    const float* __restrict__ resid) {
  const int z  = blockIdx.z;
  const int zo = z / inner, zi = z - zo * inner;
  const long long offA = (long long)zo * sAo + (long long)zi * sAi;
  const long long offB = (long long)zo * sBo + (long long)zi * sBi;
  const long long offC = (long long)zo * sCo + (long long)zi * sCi;

  const int lane = threadIdx.x & 31;
  const int w    = threadIdx.x >> 5;      // 8 waves per WG
  const int wm   = w & 1, wn = w >> 1;    // 2 (M) x 4 (N) wave grid
  const int tileM = blockIdx.y * 128 + wm * 64;
  const int tileN = blockIdx.x * 256 + wn * 64;
  if (tileM >= M || tileN >= N) return;   // all dims are multiples of 64

  const int r   = lane & 15;
  const int sel = lane >> 4;

  const unsigned short* ap[4];
  const unsigned short* bp[4];
#pragma unroll
  for (int i = 0; i < 4; ++i)
    ap[i] = A + offA + (long long)(tileM + i * 16 + r) * lda;
#pragma unroll
  for (int j = 0; j < 4; ++j)
    bp[j] = B + offB + (long long)(tileN + j * 16 + r) * ldb;

  v8f acc[4][4] = {};

  for (int kb = 0; kb < K; kb += 32) {
    v16bf af[4], bfr[4];
#pragma unroll
    for (int i = 0; i < 4; ++i) af[i] = load_a_frag(ap[i], sel);
#pragma unroll
    for (int j = 0; j < 4; ++j) bfr[j] = load_b_frag(bp[j], sel);
#pragma unroll
    for (int i = 0; i < 4; ++i)
#pragma unroll
      for (int j = 0; j < 4; ++j)
        acc[i][j] = __builtin_amdgcn_wmma_f32_16x16x32_bf16(
            false, af[i], false, bfr[j], (short)0, acc[i][j], false, false);
#pragma unroll
    for (int i = 0; i < 4; ++i) ap[i] += 32;
#pragma unroll
    for (int j = 0; j < 4; ++j) bp[j] += 32;
  }

  const GLOBAL_AS float* gbias = (const GLOBAL_AS float*)bias;
  const GLOBAL_AS float* gres  = (const GLOBAL_AS float*)resid;
  GLOBAL_AS float* gCf = (GLOBAL_AS float*)Cf;
  GLOBAL_AS unsigned short* gCb = (GLOBAL_AS unsigned short*)Cb;

  // C/D layout: lane%16 = column; VGPR v -> row v (lanes<16) / v+8 (lanes>=16)
#pragma unroll
  for (int i = 0; i < 4; ++i) {
#pragma unroll
    for (int j = 0; j < 4; ++j) {
      const int n = tileN + j * 16 + r;
      float bb = 0.f;
      if constexpr (BIAS) bb = gbias[n];
#pragma unroll
      for (int v = 0; v < 8; ++v) {
        const int m = tileM + i * 16 + sel * 8 + v;
        float x = acc[i][j][v];
        if constexpr (BIAS) x += bb;
        const long long idx = offC + (long long)m * ldc + n;
        if constexpr (RESID) x += gres[idx];
        if constexpr (ACT == 1) x = 0.5f * x * (1.0f + erff(x * 0.70710678118654752f));
        if constexpr (ACT == 2) x = 1.0f / (1.0f + __expf(-x));
        if constexpr (STF) gCf[idx] = x;
        if constexpr (STB) gCb[idx] = f2bf(x);
      }
    }
  }
}

__global__ __launch_bounds__(256)
void cvt_f32_bf16(const float* __restrict__ in, unsigned short* __restrict__ out,
                  long long n) {
  const GLOBAL_AS float* gi = (const GLOBAL_AS float*)in;
  GLOBAL_AS unsigned short* go = (GLOBAL_AS unsigned short*)out;
  long long i = (long long)blockIdx.x * 256 + threadIdx.x;
  const long long stride = (long long)gridDim.x * 256;
  for (; i < n; i += stride) go[i] = f2bf(gi[i]);
}

// LayerNorm over D per row; writes bf16 for WMMA consumption.
__global__ __launch_bounds__(256)
void layernorm_bf16(const float* __restrict__ x, const float* __restrict__ g,
                    const float* __restrict__ b, unsigned short* __restrict__ out,
                    int D) {
  const long long row = blockIdx.x;
  const GLOBAL_AS float* xr = (const GLOBAL_AS float*)(x + row * (long long)D);
  const GLOBAL_AS float* gg = (const GLOBAL_AS float*)g;
  const GLOBAL_AS float* gb = (const GLOBAL_AS float*)b;
  GLOBAL_AS unsigned short* go = (GLOBAL_AS unsigned short*)(out + row * (long long)D);
  __shared__ float ssum[256], ssq[256];
  float s = 0.f, q = 0.f;
  for (int i = threadIdx.x; i < D; i += 256) {
    const float v = xr[i];
    s += v; q += v * v;
  }
  ssum[threadIdx.x] = s; ssq[threadIdx.x] = q;
  __syncthreads();
  for (int st = 128; st > 0; st >>= 1) {
    if (threadIdx.x < st) {
      ssum[threadIdx.x] += ssum[threadIdx.x + st];
      ssq[threadIdx.x]  += ssq[threadIdx.x + st];
    }
    __syncthreads();
  }
  const float mean = ssum[0] / (float)D;
  const float var  = ssq[0] / (float)D - mean * mean;
  const float inv  = rsqrtf(var + 1e-5f);
  for (int i = threadIdx.x; i < D; i += 256) {
    go[i] = f2bf((xr[i] - mean) * inv * gg[i] + gb[i]);
  }
}

// softmax over rows of length L; fuses score scale and optional
// log(mask + 1e-6) additive bias; writes bf16 probabilities.
template <bool MASKED>
__global__ __launch_bounds__(256)
void softmax_bf16(const float* __restrict__ sc, const float* __restrict__ maskadd,
                  unsigned short* __restrict__ probs, int L, float scale) {
  const long long row = blockIdx.x;
  const GLOBAL_AS float* srow = (const GLOBAL_AS float*)(sc + row * (long long)L);
  const GLOBAL_AS float* mrow =
      MASKED ? (const GLOBAL_AS float*)(maskadd + row * (long long)L) : nullptr;
  GLOBAL_AS unsigned short* prow =
      (GLOBAL_AS unsigned short*)(probs + row * (long long)L);
  __shared__ float red[256];
  float mx = -1e30f;
  for (int i = threadIdx.x; i < L; i += 256) {
    float v = srow[i] * scale;
    if constexpr (MASKED) v += __logf(mrow[i] + 1e-6f);
    mx = fmaxf(mx, v);
  }
  red[threadIdx.x] = mx; __syncthreads();
  for (int st = 128; st > 0; st >>= 1) {
    if (threadIdx.x < st) red[threadIdx.x] = fmaxf(red[threadIdx.x], red[threadIdx.x + st]);
    __syncthreads();
  }
  mx = red[0]; __syncthreads();
  float sum = 0.f;
  for (int i = threadIdx.x; i < L; i += 256) {
    float v = srow[i] * scale;
    if constexpr (MASKED) v += __logf(mrow[i] + 1e-6f);
    sum += __expf(v - mx);
  }
  red[threadIdx.x] = sum; __syncthreads();
  for (int st = 128; st > 0; st >>= 1) {
    if (threadIdx.x < st) red[threadIdx.x] += red[threadIdx.x + st];
    __syncthreads();
  }
  const float rinv = 1.0f / red[0];
  for (int i = threadIdx.x; i < L; i += 256) {
    float v = srow[i] * scale;
    if constexpr (MASKED) v += __logf(mrow[i] + 1e-6f);
    prow[i] = f2bf(__expf(v - mx) * rinv);
  }
}

// vt_self[(b*8+h)][d][kt] = qkv[(kt*8+b)*3072 + 2048 + h*128 + d]
__global__ __launch_bounds__(256)
void vt_self_kernel(const unsigned short* __restrict__ qkv,
                    unsigned short* __restrict__ vt) {
  const long long idx = (long long)blockIdx.x * 256 + threadIdx.x;
  if (idx >= 64LL * 128 * 256) return;
  const int kt = (int)(idx & 255);
  const int d  = (int)((idx >> 8) & 127);
  const int zz = (int)(idx >> 15);
  const int b = zz >> 3, h = zz & 7;
  ((GLOBAL_AS unsigned short*)vt)[idx] =
      ((const GLOBAL_AS unsigned short*)qkv)[((long long)kt * 8 + b) * 3072 + 2048 + h * 128 + d];
}

// vt_cross[b][d][s] = Vc[(s*8+b)*1024 + d]
__global__ __launch_bounds__(256)
void vt_cross_kernel(const unsigned short* __restrict__ Vc,
                     unsigned short* __restrict__ vt) {
  const long long idx = (long long)blockIdx.x * 256 + threadIdx.x;
  if (idx >= 8LL * 1024 * 4096) return;
  const int s = (int)(idx & 4095);
  const int d = (int)((idx >> 12) & 1023);
  const int b = (int)(idx >> 22);
  ((GLOBAL_AS unsigned short*)vt)[idx] =
      ((const GLOBAL_AS unsigned short*)Vc)[((long long)s * 8 + b) * 1024 + d];
}

// ---------------------------------------------------------------------------
extern "C" void kernel_launch(void* const* d_in, const int* in_sizes, int n_in,
                              void* d_out, int out_size, void* d_ws, size_t ws_size,
                              hipStream_t stream) {
  (void)in_sizes; (void)n_in; (void)out_size; (void)ws_size;
  const long long D = 1024, NQ = 256, Bn = 8, S = 4096, FF = 4096;
  const long long TOK = NQ * Bn;      // 2048
  const long long MTOK = S * Bn;      // 32768

  const float* q_in   = (const float*)d_in[0];
  const float* mem_in = (const float*)d_in[1];
  const float* pmask  = (const float*)d_in[2];
  const float* sa_in_w = (const float*)d_in[5];
  const float* sa_in_b = (const float*)d_in[6];
  const float* sa_out_w = (const float*)d_in[7];
  const float* sa_out_b = (const float*)d_in[8];
  const float* ln1_g = (const float*)d_in[9];
  const float* ln1_b = (const float*)d_in[10];
  const float* ln2_g = (const float*)d_in[11];
  const float* ln2_b = (const float*)d_in[12];
  const float* ln3_g = (const float*)d_in[13];
  const float* ln3_b = (const float*)d_in[14];
  const float* wq = (const float*)d_in[15];
  const float* bq = (const float*)d_in[16];
  const float* wk = (const float*)d_in[17];
  const float* bk = (const float*)d_in[18];
  const float* wv = (const float*)d_in[19];
  const float* bv = (const float*)d_in[20];
  const float* wo = (const float*)d_in[21];
  const float* bo = (const float*)d_in[22];
  const float* w1 = (const float*)d_in[23];
  const float* b1 = (const float*)d_in[24];
  const float* w2 = (const float*)d_in[25];
  const float* b2 = (const float*)d_in[26];
  const float* mask_w = (const float*)d_in[27];
  const float* mask_b = (const float*)d_in[28];

  float* out_query = (float*)d_out;                  // [NQ,B,D] = 2,097,152
  float* out_mask  = (float*)d_out + 2097152;        // [B,NQ,64,64]

  // workspace allocator (256B aligned)
  char* base = (char*)d_ws;
  size_t off = 0;
  auto alloc = [&](size_t bytes) -> void* {
    void* p = base + off;
    off = (off + bytes + 255) & ~(size_t)255;
    return p;
  };
  typedef unsigned short bfu;
  bfu* mem_bf   = (bfu*)alloc(MTOK * D * 2);
  bfu* sa_in_bf = (bfu*)alloc(3 * D * D * 2);
  bfu* sa_out_bf = (bfu*)alloc(D * D * 2);
  bfu* wq_bf = (bfu*)alloc(D * D * 2);
  bfu* wk_bf = (bfu*)alloc(D * D * 2);
  bfu* wv_bf = (bfu*)alloc(D * D * 2);
  bfu* wo_bf = (bfu*)alloc(D * D * 2);
  bfu* w1_bf = (bfu*)alloc(FF * D * 2);
  bfu* w2_bf = (bfu*)alloc(D * FF * 2);
  bfu* mw_bf = (bfu*)alloc((long long)S * D * 2);
  bfu* xln     = (bfu*)alloc(TOK * D * 2);           // reused for ln1/ln2/ln3
  bfu* qkv     = (bfu*)alloc(TOK * 3 * D * 2);
  float* sc_s  = (float*)alloc(64LL * 256 * 256 * 4);
  bfu* pb_s    = (bfu*)alloc(64LL * 256 * 256 * 2);
  bfu* vt_s    = (bfu*)alloc(64LL * 128 * 256 * 2);
  bfu* attn_cat = (bfu*)alloc(TOK * D * 2);          // reused self + cross
  float* query1 = (float*)alloc(TOK * D * 4);
  bfu* Qc      = (bfu*)alloc(TOK * D * 2);
  bfu* Kc      = (bfu*)alloc(MTOK * D * 2);
  bfu* Vc      = (bfu*)alloc(MTOK * D * 2);
  bfu* VtC     = (bfu*)alloc(8LL * D * S * 2);
  float* sc_c  = (float*)alloc(8LL * 256 * 4096 * 4);
  bfu* pb_c    = (bfu*)alloc(8LL * 256 * 4096 * 2);
  float* query2 = (float*)alloc(TOK * D * 4);
  bfu* hbuf    = (bfu*)alloc(TOK * FF * 2);
  bfu* q3bf    = (bfu*)alloc(TOK * D * 2);

  auto cvt = [&](const float* in, bfu* o, long long n) {
    cvt_f32_bf16<<<dim3((unsigned)((n + 255) / 256)), dim3(256), 0, stream>>>(in, o, n);
  };
  // dispatch to the 7 epilogue instantiations actually used
  auto gemm = [&](const bfu* A, long long lda, long long sAo, long long sAi,
                  const bfu* B, long long ldb, long long sBo, long long sBi,
                  float* Cf, bfu* Cb, long long ldc, long long sCo, long long sCi,
                  int inner, int M, int N, int K,
                  const float* bias, const float* resid, int act, int batch) {
    dim3 grid((unsigned)((N + 255) / 256), (unsigned)((M + 127) / 128), (unsigned)batch);
    dim3 blk(256);
#define GO(Bi, Re, Ac, Sf, Sb)                                                 \
  gemm_bf16_wmma<Bi, Re, Ac, Sf, Sb><<<grid, blk, 0, stream>>>(                \
      A, lda, sAo, sAi, B, ldb, sBo, sBi, Cf, Cb, ldc, sCo, sCi, inner, M, N,  \
      K, bias, resid)
    const bool hb = bias, hr = resid, sf = Cf, sb = Cb;
    if      ( hb && !hr && act == 0 && !sf &&  sb) GO(true,  false, 0, false, true);
    else if (!hb && !hr && act == 0 &&  sf && !sb) GO(false, false, 0, true,  false);
    else if (!hb && !hr && act == 0 && !sf &&  sb) GO(false, false, 0, false, true);
    else if ( hb &&  hr && act == 0 &&  sf && !sb) GO(true,  true,  0, true,  false);
    else if ( hb && !hr && act == 1 && !sf &&  sb) GO(true,  false, 1, false, true);
    else if ( hb &&  hr && act == 0 &&  sf &&  sb) GO(true,  true,  0, true,  true);
    else if ( hb && !hr && act == 2 &&  sf && !sb) GO(true,  false, 2, true,  false);
#undef GO
  };

  // --- precision conversion (weights + memory) ---
  cvt(mem_in, mem_bf, MTOK * D);
  cvt(sa_in_w, sa_in_bf, 3 * D * D);
  cvt(sa_out_w, sa_out_bf, D * D);
  cvt(wq, wq_bf, D * D);  cvt(wk, wk_bf, D * D);
  cvt(wv, wv_bf, D * D);  cvt(wo, wo_bf, D * D);
  cvt(w1, w1_bf, FF * D); cvt(w2, w2_bf, D * FF);
  cvt(mask_w, mw_bf, (long long)S * D);

  // --- self attention block ---
  layernorm_bf16<<<dim3((unsigned)TOK), dim3(256), 0, stream>>>(q_in, ln1_g, ln1_b, xln, (int)D);
  // qkv = ln1(q) @ sa_in_w.T + b   -> bf16 [2048,3072]
  gemm(xln, D, 0, 0, sa_in_bf, D, 0, 0, nullptr, qkv, 3 * D, 0, 0,
       1, (int)TOK, (int)(3 * D), (int)D, sa_in_b, nullptr, 0, 1);
  // scores[b,h,q,k] = Qh @ Kh^T     (z = b*8+h)
  gemm(qkv, 24576, 3072, 128, qkv + 1024, 24576, 3072, 128,
       sc_s, nullptr, 256, 524288, 65536,
       8, 256, 256, 128, nullptr, nullptr, 0, 64);
  softmax_bf16<false><<<dim3(64 * 256), dim3(256), 0, stream>>>(
      sc_s, nullptr, pb_s, 256, 0.08838834764831845f);
  vt_self_kernel<<<dim3((unsigned)(64LL * 128 * 256 / 256)), dim3(256), 0, stream>>>(qkv, vt_s);
  // attn_out = probs @ V  -> scattered bf16 store into [token, D] head slots
  gemm(pb_s, 256, 524288, 65536, vt_s, 256, 262144, 32768,
       nullptr, attn_cat, 8192, 1024, 128,
       8, 256, 128, 256, nullptr, nullptr, 0, 64);
  // query1 = query + attn_cat @ sa_out_w.T + b
  gemm(attn_cat, D, 0, 0, sa_out_bf, D, 0, 0, query1, nullptr, D, 0, 0,
       1, (int)TOK, (int)D, (int)D, sa_out_b, q_in, 0, 1);

  // --- masked cross attention ---
  layernorm_bf16<<<dim3((unsigned)TOK), dim3(256), 0, stream>>>(query1, ln2_g, ln2_b, xln, (int)D);
  gemm(xln, D, 0, 0, wq_bf, D, 0, 0, nullptr, Qc, D, 0, 0,
       1, (int)TOK, (int)D, (int)D, bq, nullptr, 0, 1);
  gemm(mem_bf, D, 0, 0, wk_bf, D, 0, 0, nullptr, Kc, D, 0, 0,
       1, (int)MTOK, (int)D, (int)D, bk, nullptr, 0, 1);
  gemm(mem_bf, D, 0, 0, wv_bf, D, 0, 0, nullptr, Vc, D, 0, 0,
       1, (int)MTOK, (int)D, (int)D, bv, nullptr, 0, 1);
  vt_cross_kernel<<<dim3((unsigned)(8LL * 1024 * 4096 / 256)), dim3(256), 0, stream>>>(Vc, VtC);
  // scores[b,q,s] = Q @ K^T
  gemm(Qc, 8192, 1024, 0, Kc, 8192, 1024, 0,
       sc_c, nullptr, 4096, 1048576, 0,
       1, 256, 4096, 1024, nullptr, nullptr, 0, 8);
  softmax_bf16<true><<<dim3(8 * 256), dim3(256), 0, stream>>>(
      sc_c, pmask, pb_c, 4096, 0.03125f);
  // attn_out[b,q,d] -> bf16 [token, D]
  gemm(pb_c, 4096, 1048576, 0, VtC, 4096, 4194304, 0,
       nullptr, attn_cat, 8192, 1024, 0,
       1, 256, 1024, 4096, nullptr, nullptr, 0, 8);
  // query2 = query1 + attn_cat @ wo.T + bo
  gemm(attn_cat, D, 0, 0, wo_bf, D, 0, 0, query2, nullptr, D, 0, 0,
       1, (int)TOK, (int)D, (int)D, bo, query1, 0, 1);

  // --- FFN ---
  layernorm_bf16<<<dim3((unsigned)TOK), dim3(256), 0, stream>>>(query2, ln3_g, ln3_b, xln, (int)D);
  gemm(xln, D, 0, 0, w1_bf, D, 0, 0, nullptr, hbuf, FF, 0, 0,
       1, (int)TOK, (int)FF, (int)D, b1, nullptr, 1 /*GELU*/, 1);
  // query3 = query2 + h @ w2.T + b2  -> fp32 to d_out, bf16 copy for mask head
  gemm(hbuf, FF, 0, 0, w2_bf, FF, 0, 0, out_query, q3bf, D, 0, 0,
       1, (int)TOK, (int)D, (int)FF, b2, query2, 0, 1);

  // --- mask prediction: sigmoid(q3[b,q,:] @ mask_w.T + mask_b) ---
  gemm(q3bf, 8192, 1024, 0, mw_bf, D, 0, 0,
       out_mask, nullptr, 4096, 1048576, 0,
       1, 256, 4096, 1024, mask_b, nullptr, 2 /*sigmoid*/, 8);
}
